// GatedLinearAttention_10514079940651
// MI455X (gfx1250) — matrix operand order repaired and checked by reading, not compiled, not run on hardware
//
#include <hip/hip_runtime.h>
#include <hip/hip_bf16.h>

// ---------------------------------------------------------------------------
// GLA forward for MI455X (gfx1250). bf16 WMMA for all 1024x1024 projections
// (async global->LDS double-buffered B panel), fp32 VALU for the serial gated
// scan (4 timesteps per barrier phase) + LayerNorm + gates.
// ---------------------------------------------------------------------------

#define T_LEN 2048
#define B_SZ  4
#define C_DIM 1024
#define H_CNT 16
#define D_HD  64
#define ROWS  (B_SZ * T_LEN)          // 8192 rows in (B,T) major order

#define KCHUNK  64                    // K-slice staged per LDS buffer
#define BSTRIDE 80                    // padded column stride (bf16 elems) in LDS

typedef __bf16 bf16_t;
typedef __attribute__((ext_vector_type(16))) __bf16 bf16x16;
typedef __attribute__((ext_vector_type(8)))  float  f32x8;

// ---------------------------------------------------------------------------
// 1) x (T,B,C) f32 -> XB16 (B*T, C) bf16
// ---------------------------------------------------------------------------
__global__ __launch_bounds__(256) void cvt_x_kernel(const float* __restrict__ x,
                                                    bf16_t* __restrict__ xb16) {
    size_t i = (size_t)blockIdx.x * 256 + threadIdx.x;   // over 8M elements
    int c  = (int)(i & (C_DIM - 1));
    int tb = (int)(i >> 10);            // = t*B + b
    int b  = tb & (B_SZ - 1);
    int t  = tb >> 2;
    size_t orow = (size_t)b * T_LEN + t;
    xb16[orow * C_DIM + c] = (bf16_t)x[i];
}

// ---------------------------------------------------------------------------
// 2) W[K,N] f32 -> Wt[N,K] bf16 (so a B-matrix column is a contiguous row)
// ---------------------------------------------------------------------------
__global__ __launch_bounds__(256) void cvt_w_kernel(const float* __restrict__ w,
                                                    bf16_t* __restrict__ wt) {
    size_t j = (size_t)blockIdx.x * 256 + threadIdx.x;   // over 1M outputs
    int k = (int)(j & (C_DIM - 1));
    int n = (int)(j >> 10);
    wt[j] = (bf16_t)w[(size_t)k * C_DIM + n];
}

// ---------------------------------------------------------------------------
// 3) bf16 WMMA GEMM: C[M,1024] = A[M,1024] * Bt[N=1024,K=1024]^T
//    Wave tile: 32(M) x 64(N) -> 8 WMMAs per 32-K step.
//    Block = 8 waves -> 256 x 64 tile.  B panel (64 cols x 64 K) staged in LDS
//    via async global->LDS copies, double-buffered (ASYNCcnt pipelining).
//    All 4 B fragments are pre-loaded so ds waits stagger across the 8 WMMAs.
//    storeMode 0: C row-major (row = b*T+t).
//    storeMode 1: store to (t*B + b) row order (final output layout).
// ---------------------------------------------------------------------------
__global__ __launch_bounds__(256) void gemm_bf16_kernel(const bf16_t* __restrict__ A,
                                                        const bf16_t* __restrict__ Bt,
                                                        float* __restrict__ C,
                                                        int storeMode) {
    __shared__ __align__(16) bf16_t panel[2][64 * BSTRIDE];

    const int tid  = threadIdx.x;
    const int wave = tid >> 5;
    const int lane = tid & 31;
    const int lhalf = lane >> 4;
    const int lr = lane & 15;

    const int row0 = blockIdx.x * 256 + wave * 32;
    const int col0 = blockIdx.y * 64;

    f32x8 acc[2][4] = {};

    // A fragment bases (ISA 7.12.2 16-bit A layout):
    // elems 0..7  = K: kk + lhalf*8 .. +7 ; elems 8..15 = K: kk+16+lhalf*8 .. +7
    const bf16_t* pa0 = A + (size_t)(row0 + lr) * C_DIM + (lhalf ? 8 : 0);
    const bf16_t* pa1 = pa0 + (size_t)16 * C_DIM;

    // Async prefetch of a 64-col x 64-K B panel into LDS buffer `buf`.
    // 512 chunks of 16B; each thread issues 2 async copies (2 per wave in ASYNCcnt).
    auto prefetch = [&](int buf, int kc) {
#pragma unroll
        for (int r = 0; r < 2; ++r) {
            const int c   = tid + r * 256;
            const int col = c >> 3, seg = c & 7;
            const bf16_t* src = Bt + (size_t)(col0 + col) * C_DIM + kc + seg * 8;
            unsigned dst = (unsigned)(size_t)&panel[buf][col * BSTRIDE + seg * 8];
            asm volatile("global_load_async_to_lds_b128 %0, %1, off"
                         :: "v"(dst), "v"(src) : "memory");
        }
    };

    union Frag { bf16x16 v; float4 q[2]; };

    prefetch(0, 0);
    int buf = 0;
    for (int kc = 0; kc < C_DIM; kc += KCHUNK) {
        if (kc + KCHUNK < C_DIM) {
            prefetch(buf ^ 1, kc + KCHUNK);
            // async loads complete in order: <=2 outstanding => current panel landed
            asm volatile("s_wait_asynccnt 0x2" ::: "memory");
        } else {
            asm volatile("s_wait_asynccnt 0x0" ::: "memory");
        }
        __syncthreads();

#pragma unroll
        for (int ks = 0; ks < KCHUNK; ks += 32) {
            const int kk = kc + ks;
            Frag a0, a1, bfv[4];
            a0.q[0] = *(const float4*)(pa0 + kk);
            a0.q[1] = *(const float4*)(pa0 + kk + 16);
            a1.q[0] = *(const float4*)(pa1 + kk);
            a1.q[1] = *(const float4*)(pa1 + kk + 16);
#pragma unroll
            for (int nt = 0; nt < 4; ++nt) {
                // B fragment (ISA 7.12.2 16-bit B layout): lane lr = column,
                // elems 0..7 = K: ks + lhalf*8 .. +7 ; elems 8..15 = K: ks+16+lhalf*8 .. +7
                const bf16_t* pb = &panel[buf][(nt * 16 + lr) * BSTRIDE
                                              + ks + (lhalf ? 8 : 0)];
                bfv[nt].q[0] = *(const float4*)(pb);
                bfv[nt].q[1] = *(const float4*)(pb + 16);
            }
#pragma unroll
            for (int nt = 0; nt < 4; ++nt) {
                acc[0][nt] = __builtin_amdgcn_wmma_f32_16x16x32_bf16(
                    false, a0.v, false, bfv[nt].v, (short)0, acc[0][nt], false, false);
                acc[1][nt] = __builtin_amdgcn_wmma_f32_16x16x32_bf16(
                    false, a1.v, false, bfv[nt].v, (short)0, acc[1][nt], false, false);
            }
        }
        __syncthreads();   // all waves done reading before this buffer is refilled
        buf ^= 1;
    }

    // C layout (ISA 7.12.2, 32-bit C/D): elem e -> M = e + lhalf*8, N = lr
#pragma unroll
    for (int mt = 0; mt < 2; ++mt) {
#pragma unroll
        for (int nt = 0; nt < 4; ++nt) {
            const int col = col0 + nt * 16 + lr;
#pragma unroll
            for (int e = 0; e < 8; ++e) {
                const int row = row0 + mt * 16 + e + (lhalf ? 8 : 0);
                const float val = acc[mt][nt][e];
                if (storeMode == 0) {
                    C[(size_t)row * C_DIM + col] = val;
                } else {
                    const int b = row >> 11;           // row = b*T + t, T=2048
                    const int t = row & (T_LEN - 1);
                    C[((size_t)t * B_SZ + b) * C_DIM + col] = val;
                }
            }
        }
    }
}

// ---------------------------------------------------------------------------
// 4a) gate hidden: GH[row,h] = silu(x_row . Wg1[:,h] + bg1[h]), h<16
// ---------------------------------------------------------------------------
__global__ __launch_bounds__(256) void gate_hidden_kernel(const float* __restrict__ x,
                                                          const float* __restrict__ Wg1,
                                                          const float* __restrict__ bg1,
                                                          float* __restrict__ GH) {
    const int row = blockIdx.x;             // b*T + t
    const int b = row >> 11, t = row & (T_LEN - 1);
    const float* xr = x + ((size_t)t * B_SZ + b) * C_DIM;

    __shared__ float xs[C_DIM];
    const int tid = threadIdx.x;
    for (int c = tid; c < C_DIM; c += 256) xs[c] = xr[c];
    __syncthreads();

    const int h = tid & 15, chunk = tid >> 4;   // 16 h x 16 chunks
    float s = 0.f;
    const int c0 = chunk * 64;
#pragma unroll 4
    for (int c = c0; c < c0 + 64; ++c) s += xs[c] * Wg1[(size_t)c * H_CNT + h];

    __shared__ float red[16][16];
    red[h][chunk] = s;
    __syncthreads();
    for (int off = 8; off > 0; off >>= 1) {
        if (chunk < off) red[h][chunk] += red[h][chunk + off];
        __syncthreads();
    }
    if (chunk == 0) {
        float z = red[h][0] + bg1[h];
        GH[(size_t)row * H_CNT + h] = z / (1.f + expf(-z));   // silu
    }
}

// ---------------------------------------------------------------------------
// 4b) alpha[row,c] = sigmoid((GH[row,:] . Wg2[:,c] + bg2[c]) / 16)
// ---------------------------------------------------------------------------
__global__ __launch_bounds__(256) void alpha_kernel(const float* __restrict__ GH,
                                                    const float* __restrict__ Wg2,
                                                    const float* __restrict__ bg2,
                                                    float* __restrict__ ALPHA) {
    const int row = blockIdx.y;
    const int c = blockIdx.x * 256 + threadIdx.x;
    __shared__ float gh[16];
    if (threadIdx.x < 16) gh[threadIdx.x] = GH[(size_t)row * H_CNT + threadIdx.x];
    __syncthreads();
    float s = bg2[c];
#pragma unroll
    for (int h = 0; h < H_CNT; ++h) s += gh[h] * Wg2[(size_t)h * C_DIM + c];
    ALPHA[(size_t)row * C_DIM + c] = 1.f / (1.f + expf(-s * (1.f / 16.f)));
}

// ---------------------------------------------------------------------------
// 5) serial scan: one block per (b,h). S (64x64) in registers (16 f32/thread).
//    S[i,j] = a[i]*S[i,j] + k[i]*v[j];  o[j] = sum_i q[i]*S[i,j]
//    4 timesteps per barrier phase: group ig stages one of {q,k,v,a} for 4
//    steps, then 4 back-to-back register updates, then group ig reduces and
//    writes step ig's output.
// ---------------------------------------------------------------------------
__global__ __launch_bounds__(256) void scan_kernel(const float* __restrict__ Q,
                                                   const float* __restrict__ K,
                                                   const float* __restrict__ V,
                                                   const float* __restrict__ A,
                                                   float* __restrict__ OSC,
                                                   float* __restrict__ Sfinal) {
    const int bh = blockIdx.x;              // 0..63
    const int b = bh >> 4, h = bh & 15;
    const int tid = threadIdx.x;            // 256
    const int j = tid & 63, ig = tid >> 6;  // 4 row-groups of 16

    float S[16];
#pragma unroll
    for (int r = 0; r < 16; ++r) S[r] = 0.f;

    __shared__ float stage[4][4][64];       // [arr: q,k,v,a][step][j]
    __shared__ float sh[4][256];            // [step][tid] partial outputs

    const size_t base = (size_t)b * T_LEN * C_DIM + (size_t)h * D_HD;
    const float* sp = (ig == 0) ? Q : (ig == 1) ? K : (ig == 2) ? V : A;

    for (int t0 = 0; t0 < T_LEN; t0 += 4) {
        const size_t rb0 = base + (size_t)t0 * C_DIM;
        // stage 4 timesteps of one input array per thread group
#pragma unroll
        for (int s = 0; s < 4; ++s)
            stage[ig][s][j] = sp[rb0 + (size_t)s * C_DIM + j];
        __syncthreads();

#pragma unroll
        for (int s = 0; s < 4; ++s) {
            const float vj = stage[2][s][j];
            float acc = 0.f;
#pragma unroll
            for (int r = 0; r < 16; ++r) {
                const int i = ig * 16 + r;
                S[r] = stage[3][s][i] * S[r] + stage[1][s][i] * vj;
                acc += stage[0][s][i] * S[r];
            }
            sh[s][tid] = acc;
        }
        __syncthreads();

        // group ig reduces + writes the output of step s = ig
        OSC[rb0 + (size_t)ig * C_DIM + j] =
            sh[ig][j] + sh[ig][64 + j] + sh[ig][128 + j] + sh[ig][192 + j];
        __syncthreads();
    }

    // S_final: (B,H,d,d)
    float* Sf = Sfinal + (size_t)bh * D_HD * D_HD;
#pragma unroll
    for (int r = 0; r < 16; ++r) Sf[(size_t)(ig * 16 + r) * D_HD + j] = S[r];
}

// ---------------------------------------------------------------------------
// 6) per-head LayerNorm (eps=1e-5, no affine) * silu(og + bog) -> bf16 Y
// ---------------------------------------------------------------------------
__global__ __launch_bounds__(256) void post_kernel(const float* __restrict__ OSC,
                                                   const float* __restrict__ OGraw,
                                                   const float* __restrict__ bog,
                                                   bf16_t* __restrict__ Y16) {
    const int row = blockIdx.x;                 // 0..8191
    const int tid = threadIdx.x;
    const int h = tid >> 4, s = tid & 15;       // 16 lanes/head, 4 elems each

    const size_t off = (size_t)row * C_DIM + (size_t)h * D_HD + s * 4;
    const float* o = OSC + off;
    float xe[4];
    float sum = 0.f, sq = 0.f;
#pragma unroll
    for (int m = 0; m < 4; ++m) {
        xe[m] = o[m];
        sum += xe[m];
        sq  += xe[m] * xe[m];
    }

    __shared__ float r1[16][16];
    __shared__ float r2[16][16];
    r1[h][s] = sum; r2[h][s] = sq;
    __syncthreads();
    for (int offr = 8; offr > 0; offr >>= 1) {
        if (s < offr) { r1[h][s] += r1[h][s + offr]; r2[h][s] += r2[h][s + offr]; }
        __syncthreads();
    }
    const float mu  = r1[h][0] * (1.f / 64.f);
    const float var = r2[h][0] * (1.f / 64.f) - mu * mu;
    const float inv = rsqrtf(var + 1e-5f);

    const float* og = OGraw + off;
    const float* bg = bog + (size_t)h * D_HD + s * 4;
    bf16_t* y = Y16 + off;
#pragma unroll
    for (int m = 0; m < 4; ++m) {
        float g = og[m] + bg[m];
        g = g / (1.f + expf(-g));                       // silu
        y[m] = (bf16_t)(((xe[m] - mu) * inv) * g);
    }
}

// ---------------------------------------------------------------------------
// host-side launcher
// ---------------------------------------------------------------------------
extern "C" void kernel_launch(void* const* d_in, const int* in_sizes, int n_in,
                              void* d_out, int out_size, void* d_ws, size_t ws_size,
                              hipStream_t stream) {
    (void)in_sizes; (void)n_in; (void)out_size; (void)ws_size;

    const float* x   = (const float*)d_in[0];
    const float* Wq  = (const float*)d_in[1];
    const float* Wk  = (const float*)d_in[2];
    const float* Wv  = (const float*)d_in[3];
    const float* Wg1 = (const float*)d_in[4];
    const float* bg1 = (const float*)d_in[5];
    const float* Wg2 = (const float*)d_in[6];
    const float* bg2 = (const float*)d_in[7];
    const float* Wog = (const float*)d_in[8];
    const float* bog = (const float*)d_in[9];
    const float* Wo  = (const float*)d_in[10];

    float* out    = (float*)d_out;
    float* Sfinal = out + (size_t)T_LEN * B_SZ * C_DIM;   // after (T,B,C) output

    char* p = (char*)d_ws;
    auto alloc = [&](size_t bytes) -> void* {
        void* r = (void*)p;
        p += (bytes + 255) & ~(size_t)255;
        return r;
    };
    bf16_t* XB16 = (bf16_t*)alloc((size_t)ROWS * C_DIM * sizeof(bf16_t));
    bf16_t* WT[5];
    for (int i = 0; i < 5; ++i) WT[i] = (bf16_t*)alloc((size_t)C_DIM * C_DIM * sizeof(bf16_t));
    float* Qb    = (float*)alloc((size_t)ROWS * C_DIM * sizeof(float));
    float* Kb    = (float*)alloc((size_t)ROWS * C_DIM * sizeof(float));
    float* Vb    = (float*)alloc((size_t)ROWS * C_DIM * sizeof(float));
    float* ALPHA = (float*)alloc((size_t)ROWS * C_DIM * sizeof(float));
    float* OGraw = (float*)alloc((size_t)ROWS * C_DIM * sizeof(float));
    float* OSC   = (float*)alloc((size_t)ROWS * C_DIM * sizeof(float));
    float* GH    = (float*)alloc((size_t)ROWS * H_CNT * sizeof(float));
    bf16_t* Y16  = (bf16_t*)alloc((size_t)ROWS * C_DIM * sizeof(bf16_t));

    // 1) convert activations
    cvt_x_kernel<<<(ROWS * C_DIM) / 256, 256, 0, stream>>>(x, XB16);

    // 2) convert + transpose weights
    const float* Wsrc[5] = { Wq, Wk, Wv, Wog, Wo };
    for (int i = 0; i < 5; ++i)
        cvt_w_kernel<<<(C_DIM * C_DIM) / 256, 256, 0, stream>>>(Wsrc[i], WT[i]);

    // 3) WMMA projections (block tile 256x64)
    dim3 ggrid(ROWS / 256, C_DIM / 64);
    gemm_bf16_kernel<<<ggrid, 256, 0, stream>>>(XB16, WT[0], Qb, 0);
    gemm_bf16_kernel<<<ggrid, 256, 0, stream>>>(XB16, WT[1], Kb, 0);
    gemm_bf16_kernel<<<ggrid, 256, 0, stream>>>(XB16, WT[2], Vb, 0);
    gemm_bf16_kernel<<<ggrid, 256, 0, stream>>>(XB16, WT[3], OGraw, 0);

    // 4) gate MLP
    gate_hidden_kernel<<<ROWS, 256, 0, stream>>>(x, Wg1, bg1, GH);
    alpha_kernel<<<dim3(C_DIM / 256, ROWS), 256, 0, stream>>>(GH, Wg2, bg2, ALPHA);

    // 5) serial scan (writes OSC + S_final)
    scan_kernel<<<B_SZ * H_CNT, 256, 0, stream>>>(Qb, Kb, Vb, ALPHA, OSC, Sfinal);

    // 6) LayerNorm * silu gate -> bf16
    post_kernel<<<ROWS, 256, 0, stream>>>(OSC, OGraw, bog, Y16);

    // 7) final projection, stored directly in (T,B,C) order
    gemm_bf16_kernel<<<ggrid, 256, 0, stream>>>(Y16, WT[4], out, 1);
}
